// FGCN_cell_83021717831858
// MI455X (gfx1250) — matrix operand order repaired and checked by reading, not compiled, not run on hardware
//
#include <hip/hip_runtime.h>
#include <math.h>

// Problem constants
#define BB 32
#define TT 12
#define NN 1024
#define HH 64
#define CIN 65            // H+1
#define CCOLS 2080        // B*(H+1)
#define CSTRIDE 2112      // padded to 33 blocks of 64 columns

typedef __attribute__((ext_vector_type(16))) __bf16 v16bf;
typedef __attribute__((ext_vector_type(8)))  __bf16 v8bf;
typedef __attribute__((ext_vector_type(8)))  float  v8f;

__device__ __forceinline__ float sigmoidf(float x) { return 1.0f / (1.0f + __expf(-x)); }

// Issue one 16B async global->LDS transfer (ASYNCcnt-tracked DMA, gfx1250).
__device__ __forceinline__ void async_ld16(unsigned lds_off, const void* gaddr) {
    asm volatile("global_load_async_to_lds_b128 %0, %1, off"
                 :: "v"(lds_off), "v"((unsigned long long)(size_t)gaddr)
                 : "memory");
}

// Fused pipeline sync: wait own async DMA + own LDS reads, then workgroup
// barrier. Produces an SGPR token (always 0) that we fold into every LDS
// fragment address so the ds_loads carry a TRUE data dependence on the
// barrier -- the scheduler cannot hoist them above it (observed hazard in
// the previous round's schedule).
__device__ __forceinline__ unsigned sync_token() {
    unsigned t;
    asm volatile("s_wait_asynccnt 0\n\t"
                 "s_wait_dscnt 0\n\t"
                 "s_barrier_signal -1\n\t"
                 "s_barrier_wait -1\n\t"
                 "s_mov_b32 %0, 0"
                 : "=s"(t) : : "memory");
    return t;
}

// ---------------------------------------------------------------------------
// d^{-1/2} for adj (rows 0..1023 -> da) and spectrum (rows 1024..2047 -> ds)
// ---------------------------------------------------------------------------
__global__ void __launch_bounds__(256)
degree_rsqrt(const float* __restrict__ adj, const float* __restrict__ spec,
             float* __restrict__ da, float* __restrict__ ds) {
    __shared__ float red[256];
    int row = blockIdx.x;                 // 0..2047
    const float* M = (row < NN) ? adj : spec;
    int m = row & (NN - 1);
    float s = 0.0f;
    for (int n = threadIdx.x; n < NN; n += 256) s += M[(size_t)m * NN + n];
    red[threadIdx.x] = s;
    __syncthreads();
    for (int st = 128; st > 0; st >>= 1) {
        if (threadIdx.x < st) red[threadIdx.x] += red[threadIdx.x + st];
        __syncthreads();
    }
    if (threadIdx.x == 0) {
        float d = red[0];
        float v = (d > 0.0f) ? rsqrtf(d) : 0.0f;
        (row < NN ? da : ds)[m] = v;
    }
}

__global__ void __launch_bounds__(256)
zero_f32(float* __restrict__ p, unsigned n) {
    unsigned i = blockIdx.x * 256u + threadIdx.x;
    if (i < n) p[i] = 0.0f;
}

// ---------------------------------------------------------------------------
// A_t (bf16): (dtwt + L_spec + I)/2, final step (dtwt + I + L_adj + L_spec)/3
// ---------------------------------------------------------------------------
__global__ void __launch_bounds__(256)
build_A(const float* __restrict__ dtw, const float* __restrict__ adj,
        const float* __restrict__ spec, const float* __restrict__ td,
        const float* __restrict__ da, const float* __restrict__ ds,
        const int* __restrict__ srate_p, int t, int isFinal,
        __bf16* __restrict__ Aout) {
    unsigned idx = blockIdx.x * 256u + threadIdx.x;   // 0 .. N*N-1
    int m = idx >> 10, n = idx & (NN - 1);
    float srate = (float)(*srate_p);
    float tind = (float)TT - ceilf(fabsf(td[idx]) / srate);
    float w = dtw[idx];
    float dtwt = (w != 0.0f && (float)(t + 1) > tind) ? w : 0.0f;
    float eye = (m == n) ? 1.0f : 0.0f;
    float ls = eye - ds[m] * spec[idx] * ds[n];
    float val;
    if (isFinal) {
        float la = eye - da[m] * adj[idx] * da[n];
        val = (dtwt + eye + la + ls) * (1.0f / 3.0f);
    } else {
        val = (dtwt + ls + eye) * 0.5f;
    }
    Aout[idx] = (__bf16)val;
}

// ---------------------------------------------------------------------------
// Pack XT[c][n] (bf16, CSTRIDE x 1024), c = b*65: row 0 = x, rows 1..64 = h (or r*h)
// LDS-tiled transpose (64 nodes x 64 channels per block), bank-conflict padded.
// Rows 2080..2111 are zero padding (cleared once in kernel_launch).
// ---------------------------------------------------------------------------
__global__ void __launch_bounds__(256)
pack_xt(const float* __restrict__ inputs, const float* __restrict__ states,
        const float* __restrict__ gcn1, int use_r, int t,
        __bf16* __restrict__ XT) {
    __shared__ float tile[64 * 65];
    int b  = blockIdx.x >> 4;             // 0..31
    int n0 = (blockIdx.x & 15) << 6;      // 0,64,...,960
    const float* st = states + (size_t)t * BB * (NN * HH) + (size_t)b * (NN * HH) + (size_t)n0 * HH;
    const float* g1 = gcn1 + (size_t)b * (NN * 2 * HH);
#pragma unroll
    for (int it = 0; it < 16; ++it) {
        int j = it * 256 + threadIdx.x;   // 0..4095
        int nn = j >> 6, hh = j & 63;
        float v = st[j];
        if (use_r) v *= sigmoidf(g1[(size_t)(n0 + nn) * HH + hh]);
        tile[nn * 65 + hh] = v;
    }
    __syncthreads();
    size_t base = (size_t)b * CIN * NN;
#pragma unroll
    for (int it = 0; it < 16; ++it) {
        int j = it * 256 + threadIdx.x;
        int hh = j >> 6, nn = j & 63;
        XT[base + (size_t)(1 + hh) * NN + n0 + nn] = (__bf16)tile[nn * 65 + hh];
    }
    if (threadIdx.x < 64) {
        float x = inputs[(size_t)b * TT * NN + (size_t)t * NN + n0 + threadIdx.x];
        XT[base + n0 + threadIdx.x] = (__bf16)x;
    }
}

// ---------------------------------------------------------------------------
// C[1024 x CSTRIDE] f32 = A[1024x1024] bf16  x  XT^T ; WMMA 16x16x32 bf16.
// Block (8 waves) computes a 64x64 C tile. Per 32-wide K chunk, an 8KB A+B slab
// is staged global->LDS with async-LDS DMA (double buffered, overlaps WMMA).
// Wave w: M-subtile (w&3), 32-column half (w>>2) -> two 16x16 accumulators.
// Arithmetic intensity from L2: 256 KFLOP / 8KB = 32 FLOP/B.
// ---------------------------------------------------------------------------
__global__ void __launch_bounds__(256)
gemm_bf16(const __bf16* __restrict__ A, const __bf16* __restrict__ XT,
          float* __restrict__ C) {
    __shared__ __bf16 sA[2][64 * 32];     // 4KB per buffer
    __shared__ __bf16 sB[2][64 * 32];     // 4KB per buffer

    int tid  = threadIdx.x;
    int wave = tid >> 5, lane = tid & 31;
    int half = lane >> 4, l15 = lane & 15;
    int mB = blockIdx.x & 15;             // 16 M blocks
    int jB = blockIdx.x >> 4;             // 33 J blocks
    int m0 = mB << 6, j0 = jB << 6;
    int mSub = wave & 3, jSub = wave >> 2;

    // Staging assignment: thread -> row r (0..63), 8-elt segment s (0..3)
    int r = tid >> 2, sseg = tid & 3;
    const __bf16* gA = A  + (size_t)(m0 + r) * NN + sseg * 8;
    const __bf16* gB = XT + (size_t)(j0 + r) * NN + sseg * 8;
    unsigned la[2], lb[2];
    la[0] = (unsigned)(size_t)&sA[0][r * 32 + sseg * 8];
    la[1] = (unsigned)(size_t)&sA[1][r * 32 + sseg * 8];
    lb[0] = (unsigned)(size_t)&sB[0][r * 32 + sseg * 8];
    lb[1] = (unsigned)(size_t)&sB[1][r * 32 + sseg * 8];

    // Per-wave fragment byte offsets within a buffer
    unsigned offA = (unsigned)((mSub * 16 + l15) * 32 + half * 16) * sizeof(__bf16);
    unsigned offB = (unsigned)((jSub * 32 + l15) * 32 + half * 16) * sizeof(__bf16);

    v8f acc0 = {}; v8f acc1 = {};

    // Prologue: stage chunk 0, then sync (token carries the dependence)
    async_ld16(la[0], gA);
    async_ld16(lb[0], gB);
    unsigned tok = sync_token();

    for (int i = 0; i < 32; ++i) {
        int p = i & 1;
        if (i + 1 < 32) {                 // stage next chunk while computing
            int k1 = (i + 1) << 5;
            async_ld16(la[p ^ 1], gA + k1);
            async_ld16(lb[p ^ 1], gB + k1);
        }
        // LDS bases offset by the sync token: ds_loads depend on the barrier.
        const char* baseA = (const char*)sA[p] + tok;
        const char* baseB = (const char*)sB[p] + tok;
        // A fragment: row = mSub*16 + l15; K contiguous 16 per half-wave
        const __bf16* pa = (const __bf16*)(baseA + offA);
        v8bf alo = *(const v8bf*)pa;
        v8bf ahi = *(const v8bf*)(pa + 8);
        v16bf af;
#pragma unroll
        for (int q = 0; q < 8; ++q) { af[q] = alo[q]; af[q + 8] = ahi[q]; }
        // B fragments: col = jSub*32 + jt*16 + l15, K contiguous 16 per half-wave
        const __bf16* pb0 = (const __bf16*)(baseB + offB);
        const __bf16* pb1 = pb0 + 16 * 32;
        v16bf bf0 = *(const v16bf*)pb0;
        v16bf bf1 = *(const v16bf*)pb1;
        acc0 = __builtin_amdgcn_wmma_f32_16x16x32_bf16(false, af, false, bf0,
                                                       (short)0, acc0, false, false);
        acc1 = __builtin_amdgcn_wmma_f32_16x16x32_bf16(false, af, false, bf1,
                                                       (short)0, acc1, false, false);
        tok = sync_token();               // DMA landed + all waves' reads done
    }

    // C/D layout: VGPR q -> row += q + half*8, col = base + (lane&15)
    float* c0 = C + (size_t)(m0 + mSub * 16 + half * 8) * CSTRIDE
                  + j0 + jSub * 32 + l15;
#pragma unroll
    for (int q = 0; q < 8; ++q) {
        c0[(size_t)q * CSTRIDE]      = acc0[q];
        c0[(size_t)q * CSTRIDE + 16] = acc1[q];
    }
}

// ---------------------------------------------------------------------------
// gcn1[b, n*128 + c] += a1[n, b*65 + k] @ W1[k,c] + b1[c]
// ---------------------------------------------------------------------------
__global__ void __launch_bounds__(256)
conv1_acc(const float* __restrict__ Cmat, const float* __restrict__ W1,
          const float* __restrict__ b1, float* __restrict__ gcn1) {
    __shared__ float sA[2][72];
    int sub = threadIdx.x >> 7;
    int c = threadIdx.x & 127;
    int job = blockIdx.x * 2 + sub;       // 0..32767
    int b = job >> 10, n = job & (NN - 1);
    const float* arow = Cmat + (size_t)n * CSTRIDE + b * CIN;
    if (c < CIN) sA[sub][c] = arow[c];
    __syncthreads();
    float s = b1[c];
#pragma unroll 5
    for (int k = 0; k < CIN; ++k) s = fmaf(sA[sub][k], W1[k * 128 + c], s);
    gcn1[(size_t)b * (NN * 128) + (size_t)n * 128 + c] += s;
}

// ---------------------------------------------------------------------------
// gcn2 += a2 @ W2 + b2 ; on final step also h_new = u*h + (1-u)*tanh(gcn2)
// ---------------------------------------------------------------------------
__global__ void __launch_bounds__(256)
conv2_acc(const float* __restrict__ Cmat, const float* __restrict__ W2,
          const float* __restrict__ b2, const float* __restrict__ gcn1,
          const float* __restrict__ states, float* __restrict__ gcn2,
          int isFinal, float* __restrict__ out) {
    __shared__ float sA[4][68];
    int sub = threadIdx.x >> 6;
    int o = threadIdx.x & 63;
    int job = blockIdx.x * 4 + sub;       // 0..32767
    int b = job >> 10, n = job & (NN - 1);
    const float* arow = Cmat + (size_t)n * CSTRIDE + b * CIN;
    sA[sub][o] = arow[o];
    if (o == 0) sA[sub][64] = arow[64];
    __syncthreads();
    float s = b2[o];
#pragma unroll 5
    for (int k = 0; k < CIN; ++k) s = fmaf(sA[sub][k], W2[k * HH + o], s);
    size_t gi = (size_t)b * (NN * HH) + (size_t)n * HH + o;
    float g2 = gcn2[gi] + s;
    gcn2[gi] = g2;
    if (isFinal) {
        float u = sigmoidf(gcn1[(size_t)b * (NN * 2 * HH) + (size_t)(NN * HH) + (size_t)n * HH + o]);
        float h = states[(size_t)(TT - 1) * BB * (NN * HH) + gi];
        float hn = u * h + (1.0f - u) * tanhf(g2);
        out[gi] = hn;
        out[(size_t)BB * (NN * HH) + gi] = hn;   // tuple (h_new, h_new)
    }
}

// ---------------------------------------------------------------------------
extern "C" void kernel_launch(void* const* d_in, const int* in_sizes, int n_in,
                              void* d_out, int out_size, void* d_ws, size_t ws_size,
                              hipStream_t stream) {
    const float* inputs = (const float*)d_in[0];
    const float* states = (const float*)d_in[1];
    const float* dtw    = (const float*)d_in[2];
    const float* adj    = (const float*)d_in[3];
    const float* spec   = (const float*)d_in[4];
    const float* td     = (const float*)d_in[5];
    const float* W1     = (const float*)d_in[6];
    const float* b1     = (const float*)d_in[7];
    const float* W2     = (const float*)d_in[8];
    const float* b2     = (const float*)d_in[9];
    const int*   srate  = (const int*)d_in[10];
    float* out = (float*)d_out;

    char* ws = (char*)d_ws;
    size_t off = 0;
    auto take = [&](size_t bytes) {
        char* p = ws + off;
        off = (off + bytes + 255) & ~(size_t)255;
        return p;
    };
    __bf16* Abf  = (__bf16*)take((size_t)NN * NN * sizeof(__bf16));          //  2 MB
    __bf16* XT   = (__bf16*)take((size_t)CSTRIDE * NN * sizeof(__bf16));     //  4.3 MB
    float*  Cmat = (float*) take((size_t)NN * CSTRIDE * sizeof(float));      //  8.7 MB
    float*  gcn1 = (float*) take((size_t)BB * NN * 2 * HH * sizeof(float));  // 16.8 MB
    float*  gcn2 = (float*) take((size_t)BB * NN * HH * sizeof(float));      //  8.4 MB
    float*  da   = (float*) take(NN * sizeof(float));
    float*  ds   = (float*) take(NN * sizeof(float));

    degree_rsqrt<<<2 * NN, 256, 0, stream>>>(adj, spec, da, ds);

    // gcn1 and gcn2 are contiguous in ws; zero both in one pass
    unsigned nz = BB * NN * 3 * HH;
    zero_f32<<<(nz + 255) / 256, 256, 0, stream>>>(gcn1, nz);
    // zero the padding rows of XT (rows 2080..2111) once; pack never touches them
    unsigned npad = (CSTRIDE - CCOLS) * NN * sizeof(__bf16) / sizeof(float);
    zero_f32<<<(npad + 255) / 256, 256, 0, stream>>>((float*)(XT + (size_t)CCOLS * NN), npad);

    for (int t = 0; t < TT; ++t) {
        int fin = (t == TT - 1) ? 1 : 0;
        build_A<<<(NN * NN) / 256, 256, 0, stream>>>(dtw, adj, spec, td, da, ds,
                                                     srate, t, fin, Abf);
        // gconv1: cat = [x | h]
        pack_xt<<<BB * (NN / 64), 256, 0, stream>>>(inputs, states, gcn1, 0, t, XT);
        gemm_bf16<<<16 * (CSTRIDE / 64), 256, 0, stream>>>(Abf, XT, Cmat);
        conv1_acc<<<(BB * NN) / 2, 256, 0, stream>>>(Cmat, W1, b1, gcn1);
        // gconv2: cat = [x | sigmoid(gcn1_lo)*h]
        pack_xt<<<BB * (NN / 64), 256, 0, stream>>>(inputs, states, gcn1, 1, t, XT);
        gemm_bf16<<<16 * (CSTRIDE / 64), 256, 0, stream>>>(Abf, XT, Cmat);
        conv2_acc<<<(BB * NN) / 4, 256, 0, stream>>>(Cmat, W2, b2, gcn1, states,
                                                     gcn2, fin, out);
    }
}
